// Attention_21449066676376
// MI455X (gfx1250) — matrix operand order repaired and checked by reading, not compiled, not run on hardware
//
#include <hip/hip_runtime.h>
#include <hip/hip_bf16.h>

// ---------------------------------------------------------------------------
// MI455X (gfx1250) attention block, bf16 WMMA path with f32 accumulation.
//   x[8,1024,1024] @ Wqkv[1024,3072] -> Q,K,V per head (H=16, D=64)
//   flash attention per (b,h) with online softmax
//   out = (attn @ V) @ Wproj + b
// All matrix math uses v_wmma_f32_16x16x32_bf16 (wave32).
// GEMM k-loops are fully unrolled with ping-pong fragment buffers: no
// register-copy tail, constant-immediate load offsets, loads overlap WMMAs.
// ---------------------------------------------------------------------------

typedef __attribute__((ext_vector_type(16))) __bf16 bf16x16;
typedef __attribute__((ext_vector_type(8)))  __bf16 bf16x8;
typedef __attribute__((ext_vector_type(8)))  float  f32x8;

#define WMMA_BF16(A, B, C) \
  __builtin_amdgcn_wmma_f32_16x16x32_bf16(false, (A), false, (B), (short)0, (C), false, false)

static constexpr int   BATCH = 8;
static constexpr int   SEQ   = 1024;
static constexpr int   DIM   = 1024;
static constexpr int   HEADS = 16;
static constexpr int   HDIM  = 64;
static constexpr float SCALE = 0.125f;  // 64^-0.5

// ---- fragment loaders (layouts per CDNA5 ISA 7.12.2, wave32) ---------------
// A fragment: 16(M) x 32(K) bf16 from row-major A, base = &A[row0][k0].
// lane<16 : M=lane, K in {0..7, 16..23}; lane>=16: M=lane-16, K in {8..15, 24..31}
__device__ __forceinline__ bf16x16 load_a_frag(const __bf16* __restrict__ base, int ld) {
  int lane = threadIdx.x & 31;
  const __bf16* p = base + (lane & 15) * ld + (lane >> 4) * 8;
  bf16x8 lo = *(const bf16x8*)(p);        // K = 8*half .. +7
  bf16x8 hi = *(const bf16x8*)(p + 16);   // K = 16 + 8*half .. +7
  return __builtin_shufflevector(lo, hi, 0,1,2,3,4,5,6,7,8,9,10,11,12,13,14,15);
}

// B fragment: 32(K) x 16(N) bf16 where B is stored TRANSPOSED [N][K] row-major,
// base = &Bt[n0][k0].  lane holds N = lane&15, K in {8h..8h+7, 16+8h..16+8h+7}, h = lane>>4.
__device__ __forceinline__ bf16x16 load_b_frag(const __bf16* __restrict__ base, int ld) {
  int lane = threadIdx.x & 31;
  const __bf16* p = base + (lane & 15) * ld + (lane >> 4) * 8;
  bf16x8 lo = *(const bf16x8*)(p);
  bf16x8 hi = *(const bf16x8*)(p + 16);
  return __builtin_shufflevector(lo, hi, 0,1,2,3,4,5,6,7,8,9,10,11,12,13,14,15);
}

// ---- conversion kernels ----------------------------------------------------
__global__ void __launch_bounds__(256)
cvt_f32_bf16(const float* __restrict__ in, __bf16* __restrict__ out, int n) {
  int i = blockIdx.x * 256 + threadIdx.x;
  if (i < n) out[i] = (__bf16)in[i];
}

// in [K][N] f32 -> out [N][K] bf16
__global__ void __launch_bounds__(256)
cvt_transpose_bf16(const float* __restrict__ in, __bf16* __restrict__ out, int K, int N) {
  int i = blockIdx.x * 256 + threadIdx.x;
  if (i < K * N) {
    int n = i / K, kk = i - n * K;
    out[i] = (__bf16)in[(size_t)kk * N + n];
  }
}

// ---- shared GEMM core: 32(M) x 64(N) tile per wave, K = 1024 ---------------
// Fully unrolled ping-pong pipeline; all load offsets are immediates.
__device__ __forceinline__ void
gemm_core_32x64(const __bf16* __restrict__ abase, const __bf16* __restrict__ bbase,
                f32x8 acc[2][4]) {
  bf16x16 A0[2], A1[2], B[2][4];
  // phase 0 prologue at k=0
  A0[0] = load_a_frag(abase, DIM);
  A1[0] = load_a_frag(abase + (size_t)16 * DIM, DIM);
#pragma unroll
  for (int j = 0; j < 4; ++j) B[0][j] = load_b_frag(bbase + (size_t)(j * 16) * DIM, DIM);

#pragma unroll
  for (int kk = 0; kk < DIM / 64; ++kk) {
    const int ka = kk * 64 + 32;                 // odd-phase k
    A0[1] = load_a_frag(abase + ka, DIM);
    A1[1] = load_a_frag(abase + (size_t)16 * DIM + ka, DIM);
#pragma unroll
    for (int j = 0; j < 4; ++j) B[1][j] = load_b_frag(bbase + (size_t)(j * 16) * DIM + ka, DIM);

#pragma unroll
    for (int j = 0; j < 4; ++j) {
      acc[0][j] = WMMA_BF16(A0[0], B[0][j], acc[0][j]);
      acc[1][j] = WMMA_BF16(A1[0], B[0][j], acc[1][j]);
    }

    const int kb = (kk * 64 + 64) & (DIM - 1);   // even-phase k (wraps once, harmless)
    A0[0] = load_a_frag(abase + kb, DIM);
    A1[0] = load_a_frag(abase + (size_t)16 * DIM + kb, DIM);
#pragma unroll
    for (int j = 0; j < 4; ++j) B[0][j] = load_b_frag(bbase + (size_t)(j * 16) * DIM + kb, DIM);

#pragma unroll
    for (int j = 0; j < 4; ++j) {
      acc[0][j] = WMMA_BF16(A0[1], B[1][j], acc[0][j]);
      acc[1][j] = WMMA_BF16(A1[1], B[1][j], acc[1][j]);
    }
  }
}

// ---- QKV GEMM: [8192 x 1024] @ [1024 x 3072] -------------------------------
// Epilogue scatters into per-head Q (pre-scaled), K, and V-transposed layouts.
__global__ void __launch_bounds__(256)
qkv_gemm(const __bf16* __restrict__ xb, const __bf16* __restrict__ wt,
         __bf16* __restrict__ qh, __bf16* __restrict__ kh, __bf16* __restrict__ vt) {
  int wv = threadIdx.x >> 5;
  int m0 = (blockIdx.x * 8 + wv) * 32;   // [0, 8192)
  int n0 = blockIdx.y * 64;              // [0, 3072)

  f32x8 acc[2][4] = {};
  gemm_core_32x64(xb + (size_t)m0 * DIM, wt + (size_t)n0 * DIM, acc);

  int lane = threadIdx.x & 31;
  int half = lane >> 4, nl = lane & 15;
  int b_idx = m0 >> 10;          // batch (tiles never straddle batches)
  int nbase = m0 & 1023;         // seq position base
#pragma unroll
  for (int i = 0; i < 2; ++i) {
#pragma unroll
    for (int j = 0; j < 4; ++j) {
      int n = n0 + j * 16 + nl;
      int which = n >> 10;                 // 0=Q 1=K 2=V
      int h = (n & 1023) >> 6;
      int d = n & 63;
      size_t headoff = ((size_t)(b_idx * HEADS + h)) << 16;   // *SEQ*HDIM
#pragma unroll
      for (int r = 0; r < 8; ++r) {
        int m = nbase + i * 16 + r + half * 8;
        float v = acc[i][j][r];
        if (which == 0)      qh[headoff + (size_t)m * HDIM + d] = (__bf16)(v * SCALE);
        else if (which == 1) kh[headoff + (size_t)m * HDIM + d] = (__bf16)v;
        else                 vt[headoff + ((size_t)d << 10) + m] = (__bf16)v;  // [d][n]
      }
    }
  }
}

// ---- flash attention: one wave per (b, h, 16-query tile) -------------------
// K-matrix fragments prefetched one chunk ahead (loads overwrite dead regs, no
// copy tail); V fragments loaded at chunk start so softmax VALU covers latency.
__global__ void __launch_bounds__(256)
attn_flash(const __bf16* __restrict__ qh, const __bf16* __restrict__ kh,
           const __bf16* __restrict__ vt, const unsigned char* __restrict__ mask,
           __bf16* __restrict__ ao) {
  __shared__ __bf16 pshare[8][16 * 32];   // 1 KB wave-private P staging

  int wv = threadIdx.x >> 5;
  int lane = threadIdx.x & 31;
  int half = lane >> 4, nl = lane & 15;
  int gw = blockIdx.x * 8 + wv;          // 8192 waves total
  int qtile = gw & 63;
  int bh = gw >> 6;                      // [0,128)
  int b = bh >> 4;
  int h = bh & 15;
  int q0 = qtile * 16;

  const __bf16* Q = qh + ((size_t)bh << 16);
  const __bf16* K = kh + ((size_t)bh << 16);
  const __bf16* V = vt + ((size_t)bh << 16);
  __bf16* mysh = pshare[wv];

  bool mok[8];
#pragma unroll
  for (int r = 0; r < 8; ++r)
    mok[r] = mask[b * SEQ + q0 + r + half * 8] != 0;

  bf16x16 aq0 = load_a_frag(Q + (size_t)q0 * HDIM + 0, HDIM);
  bf16x16 aq1 = load_a_frag(Q + (size_t)q0 * HDIM + 32, HDIM);

  float m_i[8], l_i[8];
#pragma unroll
  for (int r = 0; r < 8; ++r) { m_i[r] = -3.0e38f; l_i[r] = 0.0f; }
  f32x8 o[4] = {};

  // prologue: K fragments for chunk 0 (kb[0..1]: keys 0..15, kb[2..3]: keys 16..31)
  bf16x16 kb[4];
  kb[0] = load_b_frag(K + 0, HDIM);
  kb[1] = load_b_frag(K + 32, HDIM);
  kb[2] = load_b_frag(K + 16 * HDIM + 0, HDIM);
  kb[3] = load_b_frag(K + 16 * HDIM + 32, HDIM);

  for (int kc = 0; kc < SEQ; kc += 32) {
    // issue V fragment loads early: latency covered by S-WMMAs + softmax VALU
    bf16x16 vb[4];
#pragma unroll
    for (int j = 0; j < 4; ++j)
      vb[j] = load_b_frag(V + (size_t)(j * 16) * SEQ + kc, SEQ);  // Vt[d][k]

    // S = Q K^T over the 32-key chunk (contraction over d = 64)
    f32x8 s0 = {}, s1 = {};
    s0 = WMMA_BF16(aq0, kb[0], s0);
    s0 = WMMA_BF16(aq1, kb[1], s0);
    s1 = WMMA_BF16(aq0, kb[2], s1);
    s1 = WMMA_BF16(aq1, kb[3], s1);

    // prefetch next chunk's K fragments (wrap -> redundant on last iter)
    int kn = (kc + 32) & (SEQ - 1);
    kb[0] = load_b_frag(K + (size_t)kn * HDIM + 0, HDIM);
    kb[1] = load_b_frag(K + (size_t)kn * HDIM + 32, HDIM);
    kb[2] = load_b_frag(K + (size_t)(kn + 16) * HDIM + 0, HDIM);
    kb[3] = load_b_frag(K + (size_t)(kn + 16) * HDIM + 32, HDIM);

    // online softmax per accumulator row (row M=r+8*half lives in one 16-lane half)
#pragma unroll
    for (int r = 0; r < 8; ++r) {
      float v0 = mok[r] ? s0[r] : -1.0e9f;
      float v1 = mok[r] ? s1[r] : -1.0e9f;
      float mx = fmaxf(v0, v1);
#pragma unroll
      for (int off = 8; off; off >>= 1) mx = fmaxf(mx, __shfl_xor(mx, off, 32));
      float mnew = fmaxf(m_i[r], mx);
      float alpha = __expf(m_i[r] - mnew);
      float p0 = __expf(v0 - mnew);
      float p1 = __expf(v1 - mnew);
      float ps = p0 + p1;
#pragma unroll
      for (int off = 8; off; off >>= 1) ps += __shfl_xor(ps, off, 32);
      l_i[r] = l_i[r] * alpha + ps;
      m_i[r] = mnew;
#pragma unroll
      for (int j = 0; j < 4; ++j) o[j][r] *= alpha;
      int row = r + half * 8;
      mysh[row * 32 + nl]      = (__bf16)p0;   // keys kc..kc+15
      mysh[row * 32 + 16 + nl] = (__bf16)p1;   // keys kc+16..kc+31
    }

    // O += P V  (contraction over 32 keys); P re-shaped via LDS into A layout
    bf16x16 ap = load_a_frag(mysh, 32);
#pragma unroll
    for (int j = 0; j < 4; ++j)
      o[j] = WMMA_BF16(ap, vb[j], o[j]);
  }

  // finalize: O /= l, store as bf16 in [b*1024+q][h*64+d] (proj GEMM A layout)
#pragma unroll
  for (int r = 0; r < 8; ++r) {
    float inv = 1.0f / l_i[r];
    int m = q0 + r + half * 8;
    size_t rowoff = ((size_t)(b * SEQ + m)) << 10;
#pragma unroll
    for (int j = 0; j < 4; ++j)
      ao[rowoff + h * HDIM + j * 16 + nl] = (__bf16)(o[j][r] * inv);
  }
}

// ---- projection GEMM with bias, f32 output ---------------------------------
__global__ void __launch_bounds__(256)
proj_gemm(const __bf16* __restrict__ ao, const __bf16* __restrict__ wt,
          const float* __restrict__ bias, float* __restrict__ out) {
  int wv = threadIdx.x >> 5;
  int m0 = (blockIdx.x * 8 + wv) * 32;   // [0, 8192)
  int n0 = blockIdx.y * 64;              // [0, 1024)

  f32x8 acc[2][4] = {};
  gemm_core_32x64(ao + (size_t)m0 * DIM, wt + (size_t)n0 * DIM, acc);

  int lane = threadIdx.x & 31;
  int half = lane >> 4, nl = lane & 15;
#pragma unroll
  for (int i = 0; i < 2; ++i) {
#pragma unroll
    for (int j = 0; j < 4; ++j) {
      int n = n0 + j * 16 + nl;
      float bb = bias[n];
#pragma unroll
      for (int r = 0; r < 8; ++r) {
        int m = m0 + i * 16 + r + half * 8;
        out[((size_t)m << 10) + n] = acc[i][j][r] + bb;
      }
    }
  }
}

// ---------------------------------------------------------------------------
extern "C" void kernel_launch(void* const* d_in, const int* in_sizes, int n_in,
                              void* d_out, int out_size, void* d_ws, size_t ws_size,
                              hipStream_t stream) {
  const float*         x      = (const float*)d_in[0];
  const unsigned char* mask   = (const unsigned char*)d_in[1];  // bool, 1 byte
  const float*         w_qkv  = (const float*)d_in[2];
  const float*         w_proj = (const float*)d_in[3];
  const float*         b_proj = (const float*)d_in[4];
  float*               out    = (float*)d_out;

  char* ws = (char*)d_ws;
  const size_t MB = 1024 * 1024;
  __bf16* xb      = (__bf16*)(ws + 0);        // 16 MB  [8192][1024]
  __bf16* wqkv_t  = (__bf16*)(ws + 16 * MB);  //  6 MB  [3072][1024]
  __bf16* wproj_t = (__bf16*)(ws + 22 * MB);  //  2 MB  [1024][1024]
  __bf16* qh      = (__bf16*)(ws + 24 * MB);  // 16 MB  [b][h][n][d], q*scale
  __bf16* kh      = (__bf16*)(ws + 40 * MB);  // 16 MB  [b][h][n][d]
  __bf16* vt      = (__bf16*)(ws + 56 * MB);  // 16 MB  [b][h][d][n]
  __bf16* ao      = (__bf16*)(ws + 72 * MB);  // 16 MB  [8192][1024]

  // 1) conversions / weight transposes
  {
    int n = BATCH * SEQ * DIM;  // 8388608
    cvt_f32_bf16<<<(n + 255) / 256, 256, 0, stream>>>(x, xb, n);
  }
  {
    int n = DIM * 3 * DIM;      // 3145728
    cvt_transpose_bf16<<<(n + 255) / 256, 256, 0, stream>>>(w_qkv, wqkv_t, DIM, 3 * DIM);
  }
  {
    int n = DIM * DIM;          // 1048576
    cvt_transpose_bf16<<<(n + 255) / 256, 256, 0, stream>>>(w_proj, wproj_t, DIM, DIM);
  }

  // 2) QKV GEMM: M=8192 (256 tiles of 32 / 8 waves -> 32), N=3072 (48 tiles of 64)
  qkv_gemm<<<dim3(32, 48), 256, 0, stream>>>(xb, wqkv_t, qh, kh, vt);

  // 3) flash attention: 8*16*64 = 8192 waves, 8 per block
  attn_flash<<<1024, 256, 0, stream>>>(qh, kh, vt, mask, ao);

  // 4) projection: M=8192, N=1024 (16 tiles of 64)
  proj_gemm<<<dim3(32, 16), 256, 0, stream>>>(ao, wproj_t, b_proj, out);
}
